// MLP_LSTM_56856777064501
// MI455X (gfx1250) — matrix operand order repaired and checked by reading, not compiled, not run on hardware
//
#include <hip/hip_runtime.h>

// ---------------- problem constants ----------------
#define Bq   512
#define Tq   512
#define Fq   52
#define EMBq 64
#define Hq   128
#define NCq  21
#define Gq   512   // 4*H

typedef __attribute__((ext_vector_type(16))) __bf16 v16bf;
typedef __attribute__((ext_vector_type(8)))  __bf16 v8bf;
typedef __attribute__((ext_vector_type(8)))  float  v8f;

// ---- A operand (16x32 bf16, row-major source buf[16][ldk]) ----
// lane<16: row M=lane, K = k0+{0..7, 16..23}; lane>=16: row M=lane-16, K = k0+{8..15, 24..31}
static __device__ __forceinline__ v16bf load_A_bf16(const __bf16* buf, int ldk, int lane, int k0) {
    int m    = lane & 15;
    int hsel = lane >> 4;
    const __bf16* p = buf + m * ldk + k0 + hsel * 8;
    v8bf lo = *(const v8bf*)p;
    v8bf hi = *(const v8bf*)(p + 16);
    return __builtin_shufflevector(lo, hi, 0,1,2,3,4,5,6,7,8,9,10,11,12,13,14,15);
}

// ---- B operand (32x16 bf16) from row-major W[N][ldk] (PyTorch layout for out = in @ W^T).
// lane<16: col N=lane, K=k0+{0..7, 16..23}; lane>=16: col N=lane-16, K=k0+{8..15, 24..31}.
static __device__ __forceinline__ v16bf load_B_bf16(const __bf16* W, int ldk, int n0, int k0, int lane) {
    int n    = lane & 15;
    int koff = (lane >> 4) * 8;
    const __bf16* p = W + (size_t)(n0 + n) * ldk + k0 + koff;
    v8bf lo = *(const v8bf*)p;
    v8bf hi = *(const v8bf*)(p + 16);
    return __builtin_shufflevector(lo, hi, 0,1,2,3,4,5,6,7,8,9,10,11,12,13,14,15);
}

static __device__ __forceinline__ v8f wmma_bf16(v16bf a, v16bf b, v8f c) {
    return __builtin_amdgcn_wmma_f32_16x16x32_bf16(false, a, false, b, (short)0, c, false, false);
}

static __device__ __forceinline__ float sigm(float x) { return 1.0f / (1.0f + __expf(-x)); }

// ---------------- weight prep: fp32 -> bf16 (+pad F 52->64, fuse b_ih+b_hh) ----------------
__global__ void prep_kernel(const float* __restrict__ W1, const float* __restrict__ W2,
                            const float* __restrict__ W_ih, const float* __restrict__ W_hh,
                            const float* __restrict__ b_ih, const float* __restrict__ b_hh,
                            __bf16* __restrict__ W1b, __bf16* __restrict__ W2b,
                            __bf16* __restrict__ W_ihb, __bf16* __restrict__ W_hhb,
                            float* __restrict__ bias_g) {
    int i = blockIdx.x * blockDim.x + threadIdx.x;
    if (i < 128 * 64) {                 // W1: [128,52] -> [128,64] padded
        int n = i >> 6, k = i & 63;
        W1b[i] = (__bf16)((k < Fq) ? W1[n * Fq + k] : 0.0f);
    }
    if (i < EMBq * 128)  W2b[i]   = (__bf16)W2[i];     // [64,128]
    if (i < Gq * EMBq)   W_ihb[i] = (__bf16)W_ih[i];   // [512,64]
    if (i < Gq * Hq)     W_hhb[i] = (__bf16)W_hh[i];   // [512,128]
    if (i < Gq)          bias_g[i] = b_ih[i] + b_hh[i];
}

// ---- fused encoder + LSTM, encoder software-pipelined one step ahead; 2 barriers/step ----
__global__ __launch_bounds__(256, 1) void mlp_lstm_kernel(
    const float* __restrict__ x,
    const __bf16* __restrict__ W1b, const __bf16* __restrict__ W2b,
    const __bf16* __restrict__ W_ihb, const __bf16* __restrict__ W_hhb,
    const float* __restrict__ b1, const float* __restrict__ b2,
    const float* __restrict__ bias_g, float* __restrict__ hn)
{
    extern __shared__ char smem[];
    __bf16* Whh_lds = (__bf16*)(smem);              // [512][128]  128 KB
    __bf16* h_tile  = (__bf16*)(smem + 131072);     // [16][128]     4 KB
    __bf16* h1_tile = (__bf16*)(smem + 135168);     // [16][128]     4 KB
    __bf16* henc_b0 = (__bf16*)(smem + 139264);     // [16][64] x2   4 KB (double buffer)
    __bf16* henc_b1 = (__bf16*)(smem + 141312);
    __bf16* x_b0    = (__bf16*)(smem + 143360);     // [16][64] x2   4 KB (double buffer)
    __bf16* x_b1    = (__bf16*)(smem + 145408);

    const int tid  = threadIdx.x;
    const int w    = tid >> 5;       // wave id 0..7
    const int lane = tid & 31;
    const int col  = lane & 15;
    const int hsel = lane >> 4;
    const int b0   = blockIdx.x * 16;

    // hoist t-invariant biases (C layout: one value per N=col)
    float b1v = b1[16 * w + col];
    float b2v = (w < 4) ? b2[16 * w + col] : 0.0f;
    float bgv[4];
#pragma unroll
    for (int q = 0; q < 4; ++q) bgv[q] = bias_g[(8 * q + w) * 16 + col];

    // stage x[:, tt, :] -> dst (pad 52->64), threads [base, 256) stride `stride`
    auto stage_x = [&](int tt, __bf16* dst, int base, int stride) {
        for (int e = base; e < 16 * 64; e += stride) {
            int i = e >> 6, j = e & 63;
            float v = 0.0f;
            if (j < Fq) v = x[((size_t)(b0 + i) * Tq + tt) * Fq + j];
            dst[e] = (__bf16)v;
        }
    };
    // GEMM1: h1 = relu(xt @ W1^T + b1), one 16x16 N-tile per wave
    auto gemm1 = [&](const __bf16* xt) {
        int n0 = 16 * w;
        v8f a;
#pragma unroll
        for (int r = 0; r < 8; ++r) a[r] = b1v;
#pragma unroll
        for (int kk = 0; kk < 2; ++kk)
            a = wmma_bf16(load_A_bf16(xt, 64, lane, 32 * kk),
                          load_B_bf16(W1b, 64, n0, 32 * kk, lane), a);
#pragma unroll
        for (int r = 0; r < 8; ++r)
            h1_tile[(r + 8 * hsel) * Hq + n0 + col] = (__bf16)fmaxf(a[r], 0.0f);
    };
    // GEMM2 (caller guards w<4): henc = relu(h1 @ W2^T + b2)
    auto gemm2 = [&](__bf16* hdst) {
        int n0 = 16 * w;
        v8f a;
#pragma unroll
        for (int r = 0; r < 8; ++r) a[r] = b2v;
#pragma unroll
        for (int kk = 0; kk < 4; ++kk)
            a = wmma_bf16(load_A_bf16(h1_tile, Hq, lane, 32 * kk),
                          load_B_bf16(W2b, Hq, n0, 32 * kk, lane), a);
#pragma unroll
        for (int r = 0; r < 8; ++r)
            hdst[(r + 8 * hsel) * EMBq + n0 + col] = (__bf16)fmaxf(a[r], 0.0f);
    };

    // ---- prologue: stage W_hh + zero h + x(0); then h1(0); then henc(0) + x(1) ----
    {
        const uint4* src = (const uint4*)W_hhb;
        uint4*       dst = (uint4*)Whh_lds;
        for (int i = tid; i < (Gq * Hq * 2) / 16; i += 256) dst[i] = src[i];
    }
    for (int i = tid; i < 16 * Hq; i += 256) h_tile[i] = (__bf16)0.0f;
    stage_x(0, x_b0, tid, 256);
    __syncthreads();

    gemm1(x_b0);
    __syncthreads();
    if (w < 4) gemm2(henc_b0);
    else       stage_x(1, x_b1, tid - 128, 128);
    __syncthreads();

    v8f cst;   // cell state, C layout: (M = r + 8*hsel, h-col = 16*w + col)
#pragma unroll
    for (int r = 0; r < 8; ++r) cst[r] = 0.0f;

    for (int t = 0; t < Tq; ++t) {
        __bf16* henc_cur = (t & 1) ? henc_b1 : henc_b0;   // henc(t)
        __bf16* henc_nxt = (t & 1) ? henc_b0 : henc_b1;   // henc(t+1) dst
        __bf16* x_nxt    = (t & 1) ? x_b0    : x_b1;      // x(t+1), staged
        __bf16* x_stage  = (t & 1) ? x_b1    : x_b0;      // x(t+2) dst

        // ================= Phase A =================
        // gates(t): A fragments shared across all 4 gate tiles -> cache in registers
        v16bf aE0 = load_A_bf16(henc_cur, EMBq, lane, 0);
        v16bf aE1 = load_A_bf16(henc_cur, EMBq, lane, 32);
        v16bf aH0 = load_A_bf16(h_tile, Hq, lane, 0);
        v16bf aH1 = load_A_bf16(h_tile, Hq, lane, 32);
        v16bf aH2 = load_A_bf16(h_tile, Hq, lane, 64);
        v16bf aH3 = load_A_bf16(h_tile, Hq, lane, 96);

        // wave w owns gate N-tiles {w, 8+w, 16+w, 24+w} => i/f/g/o for h-block w in-registers
        v8f acc[4];
#pragma unroll
        for (int q = 0; q < 4; ++q) {
            int n0 = (8 * q + w) * 16;
#pragma unroll
            for (int r = 0; r < 8; ++r) acc[q][r] = bgv[q];
            acc[q] = wmma_bf16(aE0, load_B_bf16(W_ihb, EMBq, n0, 0,  lane), acc[q]);
            acc[q] = wmma_bf16(aE1, load_B_bf16(W_ihb, EMBq, n0, 32, lane), acc[q]);
            acc[q] = wmma_bf16(aH0, load_B_bf16(Whh_lds, Hq, n0, 0,  lane), acc[q]);
            acc[q] = wmma_bf16(aH1, load_B_bf16(Whh_lds, Hq, n0, 32, lane), acc[q]);
            acc[q] = wmma_bf16(aH2, load_B_bf16(Whh_lds, Hq, n0, 64, lane), acc[q]);
            acc[q] = wmma_bf16(aH3, load_B_bf16(Whh_lds, Hq, n0, 96, lane), acc[q]);
        }
        // encoder one step ahead: h1(t+1) from staged x(t+1)
        if (t + 1 < Tq) gemm1(x_nxt);
        // stage x(t+2) into the buffer freed by GEMM1(t+1) of the previous step
        if (t + 2 < Tq) {
            stage_x(t + 2, x_stage, tid, 256);
            if (t + 3 < Tq && tid < 16)
                __builtin_prefetch(&x[((size_t)(b0 + tid) * Tq + (t + 3)) * Fq], 0, 0);
        }
        __syncthreads();

        // ================= Phase B =================
        // cell update(t): f32 VALU/trans, write new h (bf16) for next step
#pragma unroll
        for (int r = 0; r < 8; ++r) {
            float iv = sigm(acc[0][r]);
            float fv = sigm(acc[1][r]);
            float gv = tanhf(acc[2][r]);
            float ov = sigm(acc[3][r]);
            float c  = fv * cst[r] + iv * gv;
            cst[r]   = c;
            float hv = ov * tanhf(c);
            int m = r + 8 * hsel;
            h_tile[m * Hq + 16 * w + col] = (__bf16)hv;
            if (t == Tq - 1) hn[(size_t)(b0 + m) * Hq + 16 * w + col] = hv;
        }
        // encoder one step ahead: henc(t+1) from h1(t+1) (written in phase A)
        if (w < 4 && t + 1 < Tq) gemm2(henc_nxt);
        __syncthreads();
    }
}

// ---------------- classifier: out = hn @ Wc^T + bc  [512,21] ----------------
__global__ void cls_kernel(const float* __restrict__ hn, const float* __restrict__ Wc,
                           const float* __restrict__ bc, float* __restrict__ out) {
    int idx = blockIdx.x * blockDim.x + threadIdx.x;
    if (idx >= Bq * NCq) return;
    int b = idx / NCq, n = idx % NCq;
    float s = bc[n];
#pragma unroll 4
    for (int k = 0; k < Hq; ++k) s += hn[(size_t)b * Hq + k] * Wc[n * Hq + k];
    out[idx] = s;
}

extern "C" void kernel_launch(void* const* d_in, const int* in_sizes, int n_in,
                              void* d_out, int out_size, void* d_ws, size_t ws_size,
                              hipStream_t stream) {
    const float* x    = (const float*)d_in[0];
    const float* W1   = (const float*)d_in[1];
    const float* b1   = (const float*)d_in[2];
    const float* W2   = (const float*)d_in[3];
    const float* b2   = (const float*)d_in[4];
    const float* W_ih = (const float*)d_in[5];
    const float* W_hh = (const float*)d_in[6];
    const float* b_ih = (const float*)d_in[7];
    const float* b_hh = (const float*)d_in[8];
    const float* Wc   = (const float*)d_in[9];
    const float* bc   = (const float*)d_in[10];

    char* ws = (char*)d_ws;                     // ~0.5 MB total
    __bf16* W1b   = (__bf16*)(ws + 0);          // 128*64  bf16 (16 KB)
    __bf16* W2b   = (__bf16*)(ws + 16384);      // 64*128  bf16 (16 KB)
    __bf16* W_ihb = (__bf16*)(ws + 32768);      // 512*64  bf16 (64 KB)
    __bf16* W_hhb = (__bf16*)(ws + 98304);      // 512*128 bf16 (128 KB)
    float*  biasg = (float*)(ws + 229376);      // 512 f32
    float*  hn    = (float*)(ws + 231424);      // 512*128 f32 (256 KB)

    prep_kernel<<<256, 256, 0, stream>>>(W1, W2, W_ih, W_hh, b_ih, b_hh,
                                         W1b, W2b, W_ihb, W_hhb, biasg);
    mlp_lstm_kernel<<<Bq / 16, 256, 147456, stream>>>(x, W1b, W2b, W_ihb, W_hhb,
                                                      b1, b2, biasg, hn);
    cls_kernel<<<(Bq * NCq + 127) / 128, 128, 0, stream>>>(hn, Wc, bc, (float*)d_out);
}
